// NodeMoE_2963527435048
// MI455X (gfx1250) — matrix operand order repaired and checked by reading, not compiled, not run on hardware
//
#include <hip/hip_runtime.h>
#include <hip/hip_bf16.h>
#include <math.h>

// ---------------------------------------------------------------------------
// NodeMoE for MI455X (gfx1250, wave32).
// Inputs: 0:x[N,128] 1:feat[N,152] 2:noise[N,8] 3:w_gate[128,8] 4:w_noise[128,8]
//         5:W1[8,152,64] 6:b1[8,64] 7:W2[8,64,1] 8:b2[8,1]
// Output: y[N] then loss scalar.
// ---------------------------------------------------------------------------

#define N_NODES 262144
#define DIM     128
#define NEXP    8
#define HID     64
#define FDIM    152
#define ROWSTR  168           // LDS feat row stride in halfs (336B: 16B-aligned,
                              // conflict-free b128 row loads)

typedef __attribute__((ext_vector_type(16))) _Float16 v16h;
typedef __attribute__((ext_vector_type(8)))  _Float16 v8h;
typedef __attribute__((ext_vector_type(8)))  float    v8f;

#define GATES_BYTES  ((size_t)N_NODES * NEXP * 4)
#define PACKB_BYTES  ((size_t)NEXP * 4 * 5 * 32 * 16 * 2)

// Sum across each 16-lane row group, pure-VALU DPP16 butterfly.
__device__ __forceinline__ float row16_sum(float x) {
  x += __int_as_float(__builtin_amdgcn_update_dpp(0, __float_as_int(x), 0xB1, 0xF, 0xF, true));  // quad_perm xor1
  x += __int_as_float(__builtin_amdgcn_update_dpp(0, __float_as_int(x), 0x4E, 0xF, 0xF, true));  // quad_perm xor2
  x += __int_as_float(__builtin_amdgcn_update_dpp(0, __float_as_int(x), 0x141, 0xF, 0xF, true)); // row_half_mirror
  x += __int_as_float(__builtin_amdgcn_update_dpp(0, __float_as_int(x), 0x140, 0xF, 0xF, true)); // row_mirror
  return x;
}

// ---------------------------------------------------------------------------
// K0: pack W1 into per-lane WMMA B-fragment images (16-bit 32x16 B layout:
// lanes 0-15 = column N, K 0..7,16..23; lanes 16-31 K 8..15,24..31; K pairs consecutive
// per VGPR) and zero the global accumulators.
// ---------------------------------------------------------------------------
__global__ void nodemoe_prep_kernel(const float* __restrict__ W1,
                                    _Float16* __restrict__ packB,
                                    float* __restrict__ acc) {
  const int tid = blockIdx.x * blockDim.x + threadIdx.x;
  if (tid < 16) acc[tid] = 0.0f;
  if (tid >= NEXP * 4 * 5 * 32) return;
  const int lane = tid & 31;
  const int rest = tid >> 5;          // = (e*4 + t)*5 + c
  const int c    = rest % 5;
  const int et   = rest / 5;
  const int t    = et & 3;
  const int e    = et >> 2;
  const int n    = t * 16 + (lane & 15);
#pragma unroll
  for (int j = 0; j < 16; ++j) {
    const int K = c * 32 + (j & 7) + ((lane >> 4) << 3) + ((j >> 3) << 4);
    const float val = (K < FDIM) ? W1[((size_t)e * FDIM + K) * HID + n] : 0.0f;
    packB[((size_t)(rest * 32 + lane)) * 16 + j] = (_Float16)val;
  }
}

// ---------------------------------------------------------------------------
// K1: noisy top-k gating. 64 nodes / 64-thread block; x staged in LDS with
// row stride 129 floats (bank-conflict-free), weights broadcast from LDS.
// ---------------------------------------------------------------------------
__global__ __launch_bounds__(64) void nodemoe_gating_kernel(
    const float* __restrict__ x, const float* __restrict__ noise,
    const float* __restrict__ w_gate, const float* __restrict__ w_noise,
    float* __restrict__ gates, float* __restrict__ acc) {
  __shared__ float lx[64 * 129];
  __shared__ float wg[DIM * NEXP];
  __shared__ float wn[DIM * NEXP];
  __shared__ float red[16];
  const int tid = threadIdx.x;
  const int n0  = blockIdx.x * 64;

  for (int i = tid; i < DIM * NEXP; i += 64) { wg[i] = w_gate[i]; wn[i] = w_noise[i]; }
  const size_t base = (size_t)n0 * DIM;
  for (int i = tid; i < 64 * DIM; i += 64)
    lx[(i >> 7) * 129 + (i & 127)] = x[base + i];
  if (tid < 16) red[tid] = 0.0f;
  __syncthreads();

  float ag[NEXP], an[NEXP];
#pragma unroll
  for (int e = 0; e < NEXP; ++e) { ag[e] = 0.0f; an[e] = 0.0f; }
  for (int d = 0; d < DIM; ++d) {
    const float xv = lx[tid * 129 + d];
#pragma unroll
    for (int e = 0; e < NEXP; ++e) {
      ag[e] = fmaf(xv, wg[d * NEXP + e], ag[e]);
      an[e] = fmaf(xv, wn[d * NEXP + e], an[e]);
    }
  }

  const int n = n0 + tid;
  const float4 nz0 = ((const float4*)(noise + (size_t)n * NEXP))[0];
  const float4 nz1 = ((const float4*)(noise + (size_t)n * NEXP))[1];
  const float nzv[NEXP] = {nz0.x, nz0.y, nz0.z, nz0.w, nz1.x, nz1.y, nz1.z, nz1.w};

  float clean[NEXP], stdv[NEXP], noisy[NEXP];
#pragma unroll
  for (int e = 0; e < NEXP; ++e) {
    clean[e] = ag[e];
    const float z  = an[e];
    const float sp = (z > 20.0f) ? z : log1pf(expf(z));   // softplus
    stdv[e]  = sp + 0.01f;
    noisy[e] = fmaf(nzv[e], stdv[e], clean[e]);
  }

  // top-3 of 8 (K=2 selected, K+1 logits needed)
  float v1 = -1e30f; int i1 = 0;
#pragma unroll
  for (int e = 0; e < NEXP; ++e) if (noisy[e] > v1) { v1 = noisy[e]; i1 = e; }
  float v2 = -1e30f; int i2 = 0;
#pragma unroll
  for (int e = 0; e < NEXP; ++e) if (e != i1 && noisy[e] > v2) { v2 = noisy[e]; i2 = e; }
  float v3 = -1e30f;
#pragma unroll
  for (int e = 0; e < NEXP; ++e) if (e != i1 && e != i2 && noisy[e] > v3) v3 = noisy[e];

  float g[NEXP];
#pragma unroll
  for (int e = 0; e < NEXP; ++e) g[e] = 0.0f;
  const float e2  = expf(v2 - v1);
  const float inv = 1.0f / (1.0f + e2);
  g[i1] = inv;
  g[i2] = e2 * inv;

  float4* grow = (float4*)(gates + (size_t)n * NEXP);
  grow[0] = make_float4(g[0], g[1], g[2], g[3]);
  grow[1] = make_float4(g[4], g[5], g[6], g[7]);

  // smooth load estimate: Phi((clean - thr)/std); thr_in = v3, thr_out = v2
#pragma unroll
  for (int e = 0; e < NEXP; ++e) {
    const float thr = (noisy[e] > v3) ? v3 : v2;
    const float z   = (clean[e] - thr) / stdv[e];
    const float p   = 0.5f * erfcf(-z * 0.70710678118654752f);
    atomicAdd(&red[8 + e], p);
    atomicAdd(&red[e], g[e]);
  }
  __syncthreads();
  if (tid < 16) atomicAdd(&acc[tid], red[tid]);
}

// ---------------------------------------------------------------------------
// K2: expert MLP via v_wmma_f32_16x16x32_f16.
//   - 32 waves/block; wave = (expert e, hidden col-tile t); B register-resident.
//   - 64 nodes (4 WMMA node-tiles) per block iteration, grid-stride.
//   - feat staged as linear f16 rows in LDS; all 10 A-fragment b128 loads for
//     a node-tile are issued before the 5 back-to-back WMMAs (partial dscnt
//     waits instead of a full drain per WMMA).
//   - Layer-2 matvec: ReLU(h+b1)*W2 reduced across 16 lanes with a DPP16
//     butterfly (pure VALU, co-executes with the XDL pipe), b128 row stores.
// ---------------------------------------------------------------------------
__global__ __launch_bounds__(1024) void nodemoe_mlp_kernel(
    const float* __restrict__ feat, const float* __restrict__ b1,
    const float* __restrict__ W2, const float* __restrict__ b2,
    const float* __restrict__ gates, const _Float16* __restrict__ packB,
    float* __restrict__ y) {
  __shared__ alignas(16) _Float16 ldsF[64 * ROWSTR];   // 21504 B
  __shared__ float scores_part[32][64];                // [wave][node] 8 KB

  const int tid  = threadIdx.x;
  const int lane = tid & 31;
  const int w    = tid >> 5;        // 0..31
  const int e    = w >> 2;          // expert
  const int t    = w & 3;           // hidden col tile
  const int m16  = lane & 15;
  const int half = lane >> 4;       // 0|1
  const int koff = half << 3;       // upper half-wave: +8 halfs
  const int n    = t * 16 + m16;    // hidden index 0..63

  // Register-resident B fragments for this wave's (expert, col-tile).
  v16h bfrag[5];
#pragma unroll
  for (int c = 0; c < 5; ++c)
    bfrag[c] = *(const v16h*)(packB + ((size_t)(((e * 4 + t) * 5 + c) * 32 + lane)) * 16);
  const float b1v = b1[e * HID + n];
  const float w2v = W2[e * HID + n];
  float b2r[NEXP];
#pragma unroll
  for (int ee = 0; ee < NEXP; ++ee) b2r[ee] = b2[ee];

  for (int it = blockIdx.x; it < N_NODES / 64; it += gridDim.x) {
    const int n0 = it * 64;
    __syncthreads();   // protect ldsF / scores_part reuse

    // Coalesced stage: 64 feat rows -> f16 linear LDS rows, zero pad 152..159.
    {
      const float* src = feat + (size_t)n0 * FDIM;
      for (int idx = tid; idx < 64 * FDIM; idx += 1024) {
        const int m = idx / FDIM;
        const int k = idx - m * FDIM;
        ldsF[m * ROWSTR + k] = (_Float16)src[idx];
      }
      for (int i = tid; i < 64 * 8; i += 1024)
        ldsF[(i >> 3) * ROWSTR + FDIM + (i & 7)] = (_Float16)0.0f;
    }
    __syncthreads();

#pragma unroll
    for (int tt = 0; tt < 4; ++tt) {
      const int rowbase = (tt * 16 + m16) * ROWSTR;

      // Issue all A-fragment LDS loads for this node-tile up front.
      v16h afrag[5];
#pragma unroll
      for (int c = 0; c < 5; ++c) {
        const v8h lo = *(const v8h*)&ldsF[rowbase + c * 32 + koff];
        const v8h hi = *(const v8h*)&ldsF[rowbase + c * 32 + 16 + koff];
        afrag[c] = __builtin_shufflevector(
            lo, hi, 0, 1, 2, 3, 4, 5, 6, 7, 8, 9, 10, 11, 12, 13, 14, 15);
      }
      // Back-to-back WMMA chain, fp32 accumulate.
      v8f acc = {};
#pragma unroll
      for (int c = 0; c < 5; ++c)
        acc = __builtin_amdgcn_wmma_f32_16x16x32_f16(
            false, afrag[c], false, bfrag[c], (short)0, acc, false, false);

      // relu(h + b1) * W2[col]; reduce the 16 columns of this tile per row.
      float p[8];
#pragma unroll
      for (int r = 0; r < 8; ++r)
        p[r] = row16_sum(fmaxf(acc[r] + b1v, 0.0f) * w2v);
      if (m16 == 0) {
        float4* dst = (float4*)&scores_part[w][tt * 16 + (half << 3)];
        dst[0] = make_float4(p[0], p[1], p[2], p[3]);
        dst[1] = make_float4(p[4], p[5], p[6], p[7]);
      }
    }
    __syncthreads();

    // y[m] = sum_e gates[m,e] * (b2[e] + sum_t part[e,t][m])
    if (tid < 64) {
      const int m = tid;
      const float4 g0 = ((const float4*)(gates + (size_t)(n0 + m) * NEXP))[0];
      const float4 g1 = ((const float4*)(gates + (size_t)(n0 + m) * NEXP))[1];
      const float gv[NEXP] = {g0.x, g0.y, g0.z, g0.w, g1.x, g1.y, g1.z, g1.w};
      float yv = 0.0f;
#pragma unroll
      for (int ee = 0; ee < NEXP; ++ee) {
        float s = b2r[ee];
#pragma unroll
        for (int tc = 0; tc < 4; ++tc) s += scores_part[ee * 4 + tc][m];
        yv = fmaf(gv[ee], s, yv);
      }
      y[n0 + m] = yv;
    }
  }
}

// ---------------------------------------------------------------------------
// K3: loss = 0.01 * (cv2(importance) + cv2(load)), var with ddof=1.
// ---------------------------------------------------------------------------
__global__ void nodemoe_loss_kernel(const float* __restrict__ acc,
                                    float* __restrict__ out) {
  if (threadIdx.x == 0 && blockIdx.x == 0) {
    float mi = 0.0f, ml = 0.0f;
    for (int e = 0; e < NEXP; ++e) { mi += acc[e]; ml += acc[8 + e]; }
    mi *= (1.0f / NEXP); ml *= (1.0f / NEXP);
    float vi = 0.0f, vl = 0.0f;
    for (int e = 0; e < NEXP; ++e) {
      const float di = acc[e] - mi;     vi += di * di;
      const float dl = acc[8 + e] - ml; vl += dl * dl;
    }
    vi *= (1.0f / (NEXP - 1)); vl *= (1.0f / (NEXP - 1));
    out[N_NODES] = 0.01f * (vi / (mi * mi + 1e-10f) + vl / (ml * ml + 1e-10f));
  }
}

// ---------------------------------------------------------------------------
extern "C" void kernel_launch(void* const* d_in, const int* in_sizes, int n_in,
                              void* d_out, int out_size, void* d_ws, size_t ws_size,
                              hipStream_t stream) {
  const float* x       = (const float*)d_in[0];
  const float* feat    = (const float*)d_in[1];
  const float* noise   = (const float*)d_in[2];
  const float* w_gate  = (const float*)d_in[3];
  const float* w_noise = (const float*)d_in[4];
  const float* W1      = (const float*)d_in[5];
  const float* b1      = (const float*)d_in[6];
  const float* W2      = (const float*)d_in[7];
  const float* b2      = (const float*)d_in[8];
  float* out = (float*)d_out;

  char* wsb = (char*)d_ws;
  float*     gates = (float*)wsb;
  _Float16*  packB = (_Float16*)(wsb + GATES_BYTES);
  float*     acc   = (float*)(wsb + GATES_BYTES + PACKB_BYTES);

  nodemoe_prep_kernel<<<20, 256, 0, stream>>>(W1, packB, acc);
  nodemoe_gating_kernel<<<N_NODES / 64, 64, 0, stream>>>(x, noise, w_gate, w_noise,
                                                         gates, acc);
  nodemoe_mlp_kernel<<<512, 1024, 0, stream>>>(feat, b1, W2, b2, gates, packB, out);
  nodemoe_loss_kernel<<<1, 32, 0, stream>>>(acc, out);
  (void)in_sizes; (void)n_in; (void)out_size; (void)ws_size;
}
